// RFA_77841987273284
// MI455X (gfx1250) — compile-verified, matrix operand order checked by reading, not run on hardware
//
#include <hip/hip_runtime.h>
#include <hip/hip_bf16.h>
#include <math.h>

#define BB 2
#define NI 2048
#define NS 32
#define NF 16384
#define NC 4096

typedef __attribute__((ext_vector_type(16))) _Float16 v16h;
typedef __attribute__((ext_vector_type(8)))  _Float16 v8h;
typedef __attribute__((ext_vector_type(8)))  float    v8f;
typedef __attribute__((ext_vector_type(4)))  unsigned int u32x4;
typedef __attribute__((ext_vector_type(8)))  int      i32x8;
typedef __attribute__((ext_vector_type(4)))  int      i32x4;

#if defined(__has_builtin)
#if __has_builtin(__builtin_amdgcn_tensor_load_to_lds) && \
    __has_builtin(__builtin_amdgcn_s_wait_tensorcnt)
#define HAS_TDM 1
#endif
#endif

// ---------------------------------------------------------------------------
// WMMA GEMM on pre-converted f16 operands:
//   C[M,N](f32) = act(A[M,Kp](f16) @ W[Kp,Np](f16) + bias[N])
// Kp = K rounded up to 32 (zero-padded: no-op for the dot products),
// Np = N rounded up to 8. block = 256 threads (8 wave32), tile 128x64, K step 32.
// A tile staged by the Tensor Data Mover (with LDS padding done by TDM) when
// available; W tile staged transposed ([n][k]) so fragment loads are aligned
// ds_load_b128. LDS row stride 40 halves (80B, multiple of 16B).
// ---------------------------------------------------------------------------
#define TM 128
#define TN 64
#define TKK 32
#define LSTR 40

__global__ __launch_bounds__(256) void k_gemm(
    const _Float16* __restrict__ A, const _Float16* __restrict__ W,
    const float* __restrict__ bias, float* __restrict__ C,
    int M, int N, int Kp, int Np, int act)
{
  __shared__ __align__(16) _Float16 As[TM][LSTR];  // A tile, [m][k]
  __shared__ __align__(16) _Float16 Bs[TN][LSTR];  // W tile transposed, [n][k]
  const int tid  = threadIdx.x;
  const int wave = tid >> 5;
  const int lane = tid & 31;
  const int m_blk = blockIdx.y * TM;
  const int n_blk = blockIdx.x * TN;
  const int mw = m_blk + (wave << 4);

  const bool fullM  = (m_blk + TM <= M);
  const bool fullNp = (n_blk + TN <= Np);

  v8f acc[4];
  acc[0] = (v8f){}; acc[1] = (v8f){}; acc[2] = (v8f){}; acc[3] = (v8f){};

  for (int k0 = 0; k0 < Kp; k0 += TKK) {
    // ---- stage A tile (128 x 32 halves) ----
#ifdef HAS_TDM
    if (fullM) {
      if (wave == 0) {
        // TDM descriptor (ISA 8.3/8.4): 2D tile, data_size=2B,
        // tile 32(k) x 128(m), LDS pad 16B per 64B row -> 80B stride.
        unsigned lds_off = (unsigned)(uintptr_t)(&As[0][0]);
        unsigned long long ga =
            (unsigned long long)(uintptr_t)(A + (size_t)m_blk * Kp + k0);
        u32x4 g0;
        g0[0] = 1u;                                   // count=1 (valid user D#)
        g0[1] = lds_off;                              // lds_addr
        g0[2] = (unsigned)(ga & 0xFFFFFFFFu);         // global_addr[31:0]
        g0[3] = (unsigned)((ga >> 32) & 0x01FFFFFFu)  // global_addr[56:32]
              | (2u << 30);                           // type=2 ("image")
        i32x8 g1;
        g1[0] = (int)((1u << 16)    // data_size = 2 bytes
              | (1u << 20)          // pad_enable
              | (3u << 22)          // pad_interval: 16 DWORDs = 64B
              | (3u << 25));        // pad_amount: 4 DWORDs = 16B
        unsigned d0 = (unsigned)Kp;            // tensor_dim0 (elements)
        unsigned d1 = (unsigned)M;             // tensor_dim1
        g1[1] = (int)((d0 & 0xFFFFu) << 16);                       // dim0[15:0]
        g1[2] = (int)((d0 >> 16) | ((d1 & 0xFFFFu) << 16));        // dim0[31:16]|dim1[15:0]
        g1[3] = (int)((d1 >> 16) | (32u << 16));                   // dim1[31:16]|tile_dim0=32
        g1[4] = (int)128u;                                         // tile_dim1=128, tile_dim2=0
        g1[5] = (int)(unsigned)Kp;                                 // dim0_stride[31:0]
        g1[6] = 0;                                                 // dim0_stride[47:32]
        g1[7] = 0;
        i32x4 z4 = (i32x4){0, 0, 0, 0};
#if __clang_major__ >= 23
        i32x8 z8 = (i32x8){0, 0, 0, 0, 0, 0, 0, 0};
        __builtin_amdgcn_tensor_load_to_lds(g0, g1, z4, z4, z8, 0);
#else
        __builtin_amdgcn_tensor_load_to_lds(g0, g1, z4, z4, 0);
#endif
      }
    } else
#endif
    if (fullM) {
      // 512 16B chunks, 2 per thread, branch-free copies
#pragma unroll
      for (int j = 0; j < 2; ++j) {
        int i = tid + j * 256;
        int r = i >> 2, c8 = (i & 3) << 3;
        *(v8h*)(&As[r][c8]) =
            *(const v8h*)(A + (size_t)(m_blk + r) * Kp + k0 + c8);
      }
    } else {
#pragma unroll
      for (int j = 0; j < (TM * TKK) / 256; ++j) {
        int i = tid + j * 256;
        int r = i >> 5, c = i & 31;
        int gm = m_blk + r;
        As[r][c] = (gm < M) ? A[(size_t)gm * Kp + k0 + c] : (_Float16)0.0f;
      }
    }

    // ---- stage W tile (32 x 64 halves) transposed into Bs[n][k] ----
    if (fullNp) {
      // 256 16B chunks, 1 per thread
      int r = tid >> 3, c8 = (tid & 7) << 3;      // r = k, c8 = n
      v8h h = *(const v8h*)(W + (size_t)(k0 + r) * Np + n_blk + c8);
#pragma unroll
      for (int u = 0; u < 8; ++u) Bs[c8 + u][r] = h[u];
    } else {
#pragma unroll
      for (int j = 0; j < (TKK * TN) / 256; ++j) {
        int i = tid + j * 256;
        int r = i >> 6, c = i & 63;
        int gn = n_blk + c;
        Bs[c][r] = (gn < Np) ? W[(size_t)(k0 + r) * Np + gn] : (_Float16)0.0f;
      }
    }

    if (k0 + TKK < Kp) {
      int gk = k0 + TKK + (tid >> 3);
      __builtin_prefetch(&W[(size_t)gk * Np + n_blk], 0, 1);
    }
#ifdef HAS_TDM
    if (wave == 0) __builtin_amdgcn_s_wait_tensorcnt(0);
#endif
    __syncthreads();

    // A fragment, 16-bit A 16x32 layout:
    // lanes 0-15: K 0..7 / 16..23 of row M=lane; lanes 16-31: K 8..15 / 24..31
    const int arow = lane & 15;
    const int kh = (lane < 16) ? 0 : 8;
    const _Float16* ap = &As[(wave << 4) + arow][0];
    v8h alo = *(const v8h*)(ap + kh);        // ds_load_b128
    v8h ahi = *(const v8h*)(ap + kh + 16);   // ds_load_b128
    v16h af = __builtin_shufflevector(alo, ahi,
        0, 1, 2, 3, 4, 5, 6, 7, 8, 9, 10, 11, 12, 13, 14, 15);

#pragma unroll
    for (int t = 0; t < 4; ++t) {
      const _Float16* bp = &Bs[(t << 4) + arow][0];
      v8h blo = *(const v8h*)(bp + kh);
      v8h bhi = *(const v8h*)(bp + kh + 16);
      v16h bf = __builtin_shufflevector(blo, bhi,
          0, 1, 2, 3, 4, 5, 6, 7, 8, 9, 10, 11, 12, 13, 14, 15);
      acc[t] = __builtin_amdgcn_wmma_f32_16x16x32_f16(
          false, af, false, bf, (short)0, acc[t], false, false);
    }
    __syncthreads();
  }

  // epilogue: C 16x16 f32 layout (lane 0-15: N=lane, M=r; lane 16-31: N=lane-16, M=8+r)
#pragma unroll
  for (int t = 0; t < 4; ++t) {
    int col = n_blk + (t << 4) + (lane & 15);
    if (col >= N) continue;
    float bv = bias ? bias[col] : 0.0f;
#pragma unroll
    for (int r = 0; r < 8; ++r) {
      int row = mw + r + ((lane < 16) ? 0 : 8);
      if (row < M) {
        float v = acc[t][r] + bv;
        if (act) v = fmaxf(v, 0.0f);
        C[(size_t)row * N + col] = v;
      }
    }
  }
}

// f32 -> f16 with K padding: out[r][c] = c<K ? in[r][c] : 0   (out pitch Kp)
__global__ void k_cvtA(const float* __restrict__ in, _Float16* __restrict__ out,
                       long long Mr, int K, int Kp)
{
  size_t t = (size_t)blockIdx.x * blockDim.x + threadIdx.x;
  if (t >= (size_t)Mr * Kp) return;
  int c = (int)(t % Kp);
  size_t r = t / Kp;
  out[t] = (c < K) ? (_Float16)in[r * (size_t)K + c] : (_Float16)0.0f;
}

// weight f32[K][N] -> f16[Kp][Np] zero-padded
__global__ void k_cvtW(const float* __restrict__ in, _Float16* __restrict__ out,
                       int K, int N, int Kp, int Np)
{
  size_t t = (size_t)blockIdx.x * blockDim.x + threadIdx.x;
  if (t >= (size_t)Kp * Np) return;
  int n = (int)(t % Np);
  int k = (int)(t / Np);
  out[t] = (k < K && n < N) ? (_Float16)in[(size_t)k * N + n] : (_Float16)0.0f;
}

// ---------------------------------------------------------------------------
// Farthest point sampling: one block per batch, dists scratch in global ws
// ---------------------------------------------------------------------------
__global__ void k_fps(const float* __restrict__ xyz, int* __restrict__ out,
                      float* __restrict__ dists, int N, int np)
{
  __shared__ float rv[256];
  __shared__ int   ri[256];
  const int b = blockIdx.x, tid = threadIdx.x, bd = blockDim.x;
  const float* p = xyz + (size_t)b * N * 3;
  int* o = out + (size_t)b * np;
  float* dd = dists + (size_t)b * N;
  for (int i = tid; i < N; i += bd) dd[i] = 1e10f;
  if (tid == 0) o[0] = 0;
  __syncthreads();
  int last = 0;
  for (int s = 1; s < np; ++s) {
    float lx = p[last*3+0], ly = p[last*3+1], lz = p[last*3+2];
    float best = -1.0f; int bi = 0;
    for (int i = tid; i < N; i += bd) {
      float dx = p[i*3+0]-lx, dy = p[i*3+1]-ly, dz = p[i*3+2]-lz;
      float d = dx*dx + dy*dy + dz*dz;
      float m = fminf(dd[i], d);
      dd[i] = m;
      if (m > best) { best = m; bi = i; }
    }
    rv[tid] = best; ri[tid] = bi;
    __syncthreads();
    for (int off = bd >> 1; off > 0; off >>= 1) {
      if (tid < off && rv[tid + off] > rv[tid]) {
        rv[tid] = rv[tid + off]; ri[tid] = ri[tid + off];
      }
      __syncthreads();
    }
    last = ri[0];
    if (tid == 0) o[s] = last;
    __syncthreads();
  }
}

// out[b,j,c] = in[b, idx[b,j], c]
__global__ void k_gather_rows(const float* __restrict__ in, const int* __restrict__ idx,
                              float* __restrict__ out, int Nin, int J, int C, int Bn)
{
  size_t t = (size_t)blockIdx.x * blockDim.x + threadIdx.x;
  size_t total = (size_t)Bn * J * C;
  if (t >= total) return;
  int c = (int)(t % C);
  size_t r = t / C;
  int j = (int)(r % J);
  int b = (int)(r / J);
  int pi = idx[(size_t)b * J + j];
  out[t] = in[((size_t)b * Nin + pi) * C + c];
}

__global__ void k_ball(const float* __restrict__ xyz, const float* __restrict__ q,
                       int* __restrict__ out, int N, int Q, float r2, int ns, int Bn)
{
  size_t t = (size_t)blockIdx.x * blockDim.x + threadIdx.x;
  if (t >= (size_t)Bn * Q) return;
  int b = (int)(t / Q), qi = (int)(t % Q);
  const float* p = xyz + (size_t)b * N * 3;
  const float* c = q + ((size_t)b * Q + qi) * 3;
  int* o = out + ((size_t)b * Q + qi) * ns;
  int cnt = 0, first = 0; bool haveFirst = false;
  float cx = c[0], cy = c[1], cz = c[2];
  for (int i = 0; i < N && cnt < ns; ++i) {
    float dx = p[i*3+0]-cx, dy = p[i*3+1]-cy, dz = p[i*3+2]-cz;
    float d = dx*dx + dy*dy + dz*dz;
    if (d < r2) {
      if (!haveFirst) { first = i; haveFirst = true; }
      o[cnt++] = i;
    }
  }
  for (; cnt < ns; ++cnt) o[cnt] = first;
}

// g[(b,q,s), col] = col<3 ? xyz[b,idx]-new_xyz[b,q] : feats[b,idx,col-3]
__global__ void k_group(const float* __restrict__ xyz, const float* __restrict__ feats,
                        const float* __restrict__ q, const int* __restrict__ bidx,
                        float* __restrict__ g, int N, int Q, int ns, int Cf, int Bn)
{
  size_t t = (size_t)blockIdx.x * blockDim.x + threadIdx.x;
  int Cg = 3 + Cf;
  size_t total = (size_t)Bn * Q * ns * Cg;
  if (t >= total) return;
  int col = (int)(t % Cg);
  size_t row = t / Cg;
  int s = (int)(row % ns);
  size_t qb = row / ns;
  int qi = (int)(qb % Q);
  int b = (int)(qb / Q);
  int pi = bidx[((size_t)b * Q + qi) * ns + s];
  float v;
  if (col < 3)
    v = xyz[((size_t)b * N + pi) * 3 + col] - q[((size_t)b * Q + qi) * 3 + col];
  else
    v = feats[((size_t)b * N + pi) * Cf + (col - 3)];
  g[row * Cg + col] = v;
}

// out[g,c] = max_s in[(g*ns+s), c]
__global__ void k_maxpool(const float* __restrict__ in, float* __restrict__ out,
                          int groups, int ns, int C)
{
  size_t t = (size_t)blockIdx.x * blockDim.x + threadIdx.x;
  if (t >= (size_t)groups * C) return;
  int c = (int)(t % C);
  size_t g = t / C;
  float m = -3.4e38f;
  for (int s = 0; s < ns; ++s)
    m = fmaxf(m, in[((size_t)g * ns + s) * C + c]);
  out[t] = m;
}

// 3-NN inverse-distance interpolation
__global__ void k_interp3(const float* __restrict__ ux, const float* __restrict__ kxyz,
                          const float* __restrict__ kf, float* __restrict__ out,
                          int Nn, int Nk, int C, int Bn)
{
  size_t t = (size_t)blockIdx.x * blockDim.x + threadIdx.x;
  if (t >= (size_t)Bn * Nn) return;
  int b = (int)(t / Nn), n = (int)(t % Nn);
  const float* u = ux + ((size_t)b * Nn + n) * 3;
  float uxx = u[0], uyy = u[1], uzz = u[2];
  float d0 = 1e30f, d1 = 1e30f, d2 = 1e30f;
  int i0 = 0, i1 = 0, i2 = 0;
  const float* kp = kxyz + (size_t)b * Nk * 3;
  for (int i = 0; i < Nk; ++i) {
    float dx = kp[i*3+0]-uxx, dy = kp[i*3+1]-uyy, dz = kp[i*3+2]-uzz;
    float d = dx*dx + dy*dy + dz*dz;
    if (d < d0)      { d2=d1; i2=i1; d1=d0; i1=i0; d0=d; i0=i; }
    else if (d < d1) { d2=d1; i2=i1; d1=d;  i1=i; }
    else if (d < d2) { d2=d;  i2=i; }
  }
  float w0 = 1.0f/(d0+1e-8f), w1 = 1.0f/(d1+1e-8f), w2 = 1.0f/(d2+1e-8f);
  float ws = w0 + w1 + w2;
  w0 /= ws; w1 /= ws; w2 /= ws;
  float* o = out + t * (size_t)C;
  const float* f0 = kf + ((size_t)b * Nk + i0) * C;
  const float* f1 = kf + ((size_t)b * Nk + i1) * C;
  const float* f2 = kf + ((size_t)b * Nk + i2) * C;
  for (int c = 0; c < C; ++c)
    o[c] = w0 * f0[c] + w1 * f1[c] + w2 * f2[c];
}

__global__ void k_copy_cols(float* __restrict__ dst, int Cd, int co,
                            const float* __restrict__ src, int Cs, long long R)
{
  size_t t = (size_t)blockIdx.x * blockDim.x + threadIdx.x;
  if (t >= (size_t)R * Cs) return;
  int c = (int)(t % Cs);
  size_t r = t / Cs;
  dst[r * Cd + co + c] = src[r * Cs + c];
}

__global__ void k_diff_cols(float* __restrict__ dst, int Cd, int co,
                            const float* __restrict__ a, const float* __restrict__ b,
                            int Cs, long long R)
{
  size_t t = (size_t)blockIdx.x * blockDim.x + threadIdx.x;
  if (t >= (size_t)R * Cs) return;
  int c = (int)(t % Cs);
  size_t r = t / Cs;
  dst[r * Cd + co + c] = a[r * Cs + c] - b[r * Cs + c];
}

__global__ void k_broadcast(const float* __restrict__ row, float* __restrict__ out,
                            int Npts, int C, int Bn)
{
  size_t t = (size_t)blockIdx.x * blockDim.x + threadIdx.x;
  if (t >= (size_t)Bn * Npts * C) return;
  int c = (int)(t % C);
  int b = (int)(t / ((size_t)Npts * C));
  out[t] = row[(size_t)b * C + c];
}

// xt[b,n,e] = sum_d x[b,n,d] * (Tr[b,3d+e] + I)
__global__ void k_xt(const float* __restrict__ x, const float* __restrict__ Tr,
                     float* __restrict__ out, int Npts, int Bn)
{
  size_t t = (size_t)blockIdx.x * blockDim.x + threadIdx.x;
  if (t >= (size_t)Bn * Npts * 3) return;
  int e = (int)(t % 3);
  size_t r = t / 3;
  int n = (int)(r % Npts);
  int b = (int)(r / Npts);
  const float* xp = x + ((size_t)b * Npts + n) * 3;
  const float* Tp = Tr + (size_t)b * 9;
  float acc = 0.0f;
  for (int d = 0; d < 3; ++d)
    acc += xp[d] * (Tp[d*3 + e] + ((d == e) ? 1.0f : 0.0f));
  out[t] = acc;
}

__global__ void k_scatter_branch(const float* __restrict__ src, float* __restrict__ dst,
                                 int branch, int Npts, int C, int Bn)
{
  size_t t = (size_t)blockIdx.x * blockDim.x + threadIdx.x;
  if (t >= (size_t)Bn * Npts * C) return;
  int c = (int)(t % C);
  size_t r = t / C;
  int n = (int)(r % Npts);
  int b = (int)(r / Npts);
  size_t drow = (size_t)b * (8 * (size_t)Npts) + (size_t)branch * Npts + n;
  dst[drow * C + c] = src[t];
}

__global__ void k_softplus(const float* __restrict__ in, float* __restrict__ out, long long n)
{
  size_t t = (size_t)blockIdx.x * blockDim.x + threadIdx.x;
  if (t >= (size_t)n) return;
  float v = in[t];
  out[t] = fmaxf(v, 0.0f) + log1pf(expf(-fabsf(v)));
}

// per-batch bitonic sort (desc) of (score, idx) in LDS; emit first K indices
__global__ void k_topk(const float* __restrict__ score, int* __restrict__ tidx, int N, int K)
{
  extern __shared__ unsigned char sraw[];
  float* sv = (float*)sraw;
  int*   si = (int*)(sv + N);
  const int b = blockIdx.x, tid = threadIdx.x, bd = blockDim.x;
  for (int i = tid; i < N; i += bd) { sv[i] = score[(size_t)b * N + i]; si[i] = i; }
  __syncthreads();
  for (int k = 2; k <= N; k <<= 1) {
    for (int j = k >> 1; j > 0; j >>= 1) {
      for (int i = tid; i < N; i += bd) {
        int ixj = i ^ j;
        if (ixj > i) {
          bool descBlock = ((i & k) == 0);
          float a = sv[i], bb2 = sv[ixj];
          bool doSwap = descBlock ? (a < bb2) : (a > bb2);
          if (doSwap) {
            sv[i] = bb2; sv[ixj] = a;
            int ti = si[i]; si[i] = si[ixj]; si[ixj] = ti;
          }
        }
      }
      __syncthreads();
    }
  }
  for (int i = tid; i < K; i += bd) tidx[(size_t)b * K + i] = si[i];
}

__global__ void k_repeat(const float* __restrict__ in, float* __restrict__ out,
                         int J, int f, int C, int Bn)
{
  size_t t = (size_t)blockIdx.x * blockDim.x + threadIdx.x;
  if (t >= (size_t)Bn * J * f * C) return;
  int c = (int)(t % C);
  size_t r = t / C;
  int jf = (int)(r % ((size_t)J * f));
  int b = (int)(r / ((size_t)J * f));
  out[t] = in[((size_t)b * J + jf / f) * C + c];
}

// fin cols: [0,2) grid seed, [2,5) center, [5,69) cfeat_r, [69,1093) gl[b]
__global__ void k_build_fin(float* __restrict__ fin, const float* __restrict__ center,
                            const float* __restrict__ cfr, const float* __restrict__ gl,
                            int Npts, int Bn)
{
  size_t t = (size_t)blockIdx.x * blockDim.x + threadIdx.x;
  size_t total = (size_t)Bn * Npts * 1093;
  if (t >= total) return;
  int col = (int)(t % 1093);
  size_t row = t / 1093;
  int n = (int)(row % Npts);
  int b = (int)(row / Npts);
  float v;
  if (col < 2) {
    int s = n & 3;
    int sel = (col == 0) ? (s & 1) : (s >> 1);
    v = sel ? 0.05f : -0.05f;
  } else if (col < 5) {
    v = center[row * 3 + (col - 2)];
  } else if (col < 69) {
    v = cfr[row * 64 + (col - 5)];
  } else {
    v = gl[(size_t)b * 1024 + (col - 69)];
  }
  fin[t] = v;
}

__global__ void k_copyf(const float* __restrict__ a, float* __restrict__ c, long long n)
{
  size_t t = (size_t)blockIdx.x * blockDim.x + threadIdx.x;
  if (t < (size_t)n) c[t] = a[t];
}

__global__ void k_addf(const float* __restrict__ a, const float* __restrict__ b,
                       float* __restrict__ c, long long n)
{
  size_t t = (size_t)blockIdx.x * blockDim.x + threadIdx.x;
  if (t < (size_t)n) c[t] = a[t] + b[t];
}

// ---------------------------------------------------------------------------
// Host orchestration
// ---------------------------------------------------------------------------
extern "C" void kernel_launch(void* const* d_in, const int* in_sizes, int n_in,
                              void* d_out, int out_size, void* d_ws, size_t ws_size,
                              hipStream_t stream)
{
  (void)out_size; (void)ws_size;
  // locate x by its unique flat size B*NI*3
  int xi = 0;
  for (int i = 0; i < n_in; ++i) if (in_sizes[i] == BB * NI * 3) { xi = i; break; }
  const float* x = (const float*)d_in[xi];
  const float* P[200]; int npar = 0;
  for (int i = 0; i < n_in && npar < 200; ++i)
    if (i != xi) P[npar++] = (const float*)d_in[i];

  struct Layer { const float* W; const float* b; int cin, cout; };
  struct Mlp { Layer l[3]; int n; };
  int cur = 0;
  auto takeMlp = [&](const int* dims, int nl) -> Mlp {
    Mlp m; m.n = nl;
    for (int i = 0; i < nl; ++i) {
      m.l[i].W = P[cur++]; m.l[i].b = P[cur++];
      m.l[i].cin = dims[i]; m.l[i].cout = dims[i + 1];
    }
    return m;
  };

  // leaf order: jax-style sorted dict keys
  const int dfin[4] = {1093,512,512,3};
  const int dfp1[2] = {1024,64}, dfp2[2] = {1024,64}, dfp3[2] = {512,64},
            dfp4[2] = {256,64}, dfp5[2] = {128,64};
  const int dgsa[4] = {1027,512,512,1024};
  const int dmlp[3] = {387,256,128};
  const int dsa1[4] = {6,32,32,64}, dsa2[4] = {67,64,64,128}, dsa3[4] = {131,128,128,256},
            dsa4[4] = {259,256,256,512}, dsa5[4] = {515,512,512,1024};
  const int dsco[4] = {64,16,8,1};
  const int dsm1[2] = {128,64}, dsm2[2] = {64,3};
  const int dstc[4] = {3,64,128,1024}, dstf[4] = {1024,512,256,9};

  Mlp m_final = takeMlp(dfin, 3);
  Mlp m_fp1 = takeMlp(dfp1, 1), m_fp2 = takeMlp(dfp2, 1), m_fp3 = takeMlp(dfp3, 1),
      m_fp4 = takeMlp(dfp4, 1), m_fp5 = takeMlp(dfp5, 1);
  Mlp m_gsa = takeMlp(dgsa, 3);
  Mlp m_mlps[8];
  for (int i = 0; i < 8; ++i) m_mlps[i] = takeMlp(dmlp, 2);
  Mlp m_sa1 = takeMlp(dsa1, 3), m_sa2 = takeMlp(dsa2, 3), m_sa3 = takeMlp(dsa3, 3),
      m_sa4 = takeMlp(dsa4, 3), m_sa5 = takeMlp(dsa5, 3);
  Mlp m_score = takeMlp(dsco, 3);
  Mlp m_sm1 = takeMlp(dsm1, 1), m_sm2 = takeMlp(dsm2, 1);
  Mlp m_stnc[4], m_stnf[4];
  for (int i = 0; i < 4; ++i) { m_stnc[i] = takeMlp(dstc, 3); m_stnf[i] = takeMlp(dstf, 3); }

  // workspace bump allocator
  char* wsb = (char*)d_ws;
  size_t off = 0;
  auto alloc = [&](size_t bytes) -> void* {
    void* p = wsb + off;
    off = (off + bytes + 255) & ~(size_t)255;
    return p;
  };
  auto allocF = [&](size_t n) -> float* { return (float*)alloc(n * 4); };
  auto allocI = [&](size_t n) -> int*   { return (int*)alloc(n * 4); };
  auto allocH = [&](size_t n) -> _Float16* { return (_Float16*)alloc(n * 2); };

  const size_t TSZ = (size_t)18 * 1024 * 1024;  // floats per fp32 transient
  float* T1 = allocF(TSZ);
  float* T2 = allocF(TSZ);
  float* T3 = allocF(TSZ);
  _Float16* HA = allocH((size_t)38 * 1024 * 1024);  // f16 activation staging
  _Float16* HW = allocH((size_t)1  * 1024 * 1024);  // f16 weight staging

  auto G1 = [&](size_t n) { return dim3((unsigned)((n + 255) / 256)); };
  auto gemm = [&](const float* A, const Layer& L, float* C, size_t M, int act) {
    int Kp = (L.cin + 31) & ~31;
    int Np = (L.cout + 7) & ~7;
    k_cvtA<<<G1(M * (size_t)Kp), dim3(256), 0, stream>>>(A, HA, (long long)M, L.cin, Kp);
    k_cvtW<<<G1((size_t)Kp * Np), dim3(256), 0, stream>>>(L.W, HW, L.cin, L.cout, Kp, Np);
    dim3 g((L.cout + TN - 1) / TN, (unsigned)((M + TM - 1) / TM)), blk(256);
    k_gemm<<<g, blk, 0, stream>>>(HA, HW, L.b, C, (int)M, L.cout, Kp, Np, act);
  };
  auto run_mlp = [&](const Mlp& m, const float* in, size_t M, float* pA, float* pB,
                     int lastAct) -> float* {
    const float* c = in; float* dst = pA;
    for (int i = 0; i < m.n; ++i) {
      dst = (i & 1) ? pB : pA;
      int act = (lastAct || i < m.n - 1) ? 1 : 0;
      gemm(c, m.l[i], dst, M, act);
      c = dst;
    }
    return dst;
  };

  auto sa = [&](const float* xyz, const float* feats, int N, int Cf, const Mlp& m,
                int Q, float radius, float* new_xyz, float* out_f) {
    int* fi = allocI((size_t)BB * Q);
    float* dw = allocF((size_t)BB * N);
    k_fps<<<dim3(BB), dim3(256), 0, stream>>>(xyz, fi, dw, N, Q);
    k_gather_rows<<<G1((size_t)BB * Q * 3), dim3(256), 0, stream>>>(xyz, fi, new_xyz, N, Q, 3, BB);
    int* bi = allocI((size_t)BB * Q * NS);
    k_ball<<<G1((size_t)BB * Q), dim3(256), 0, stream>>>(xyz, new_xyz, bi, N, Q,
                                                         radius * radius, NS, BB);
    int Cg = 3 + Cf;
    size_t R = (size_t)BB * Q * NS;
    k_group<<<G1(R * Cg), dim3(256), 0, stream>>>(xyz, feats, new_xyz, bi, T1, N, Q, NS, Cf, BB);
    float* res = run_mlp(m, T1, R, T2, T3, 1);
    int Cout = m.l[m.n - 1].cout;
    k_maxpool<<<G1((size_t)BB * Q * Cout), dim3(256), 0, stream>>>(res, out_f, BB * Q, NS, Cout);
  };

  // ---- SA stack ----
  float* l1_xyz = allocF((size_t)BB * NI * 3);   float* l1_f = allocF((size_t)BB * NI * 64);
  float* l2_xyz = allocF((size_t)BB * 1024 * 3); float* l2_f = allocF((size_t)BB * 1024 * 128);
  float* l3_xyz = allocF((size_t)BB * 512 * 3);  float* l3_f = allocF((size_t)BB * 512 * 256);
  float* l4_xyz = allocF((size_t)BB * 256 * 3);  float* l4_f = allocF((size_t)BB * 256 * 512);
  float* l5_xyz = allocF((size_t)BB * 128 * 3);  float* l5_f = allocF((size_t)BB * 128 * 1024);

  sa(x,      x,    NI,   3,   m_sa1, NI,   0.05f, l1_xyz, l1_f);
  sa(l1_xyz, l1_f, NI,   64,  m_sa2, 1024, 0.10f, l2_xyz, l2_f);
  sa(l2_xyz, l2_f, 1024, 128, m_sa3, 512,  0.20f, l3_xyz, l3_f);
  sa(l3_xyz, l3_f, 512,  256, m_sa4, 256,  0.30f, l4_xyz, l4_f);
  sa(l4_xyz, l4_f, 256,  512, m_sa5, 128,  0.40f, l5_xyz, l5_f);

  // ---- group_all -> gl ----
  {
    long long R = (long long)BB * 128;
    k_copy_cols<<<G1((size_t)R * 3), dim3(256), 0, stream>>>(T1, 1027, 0, l5_xyz, 3, R);
    k_copy_cols<<<G1((size_t)R * 1024), dim3(256), 0, stream>>>(T1, 1027, 3, l5_f, 1024, R);
  }
  float* gres = run_mlp(m_gsa, T1, (size_t)BB * 128, T2, T3, 1);
  float* gl = allocF((size_t)BB * 1024);
  k_maxpool<<<G1((size_t)BB * 1024), dim3(256), 0, stream>>>(gres, gl, BB, 128, 1024);

  // ---- up_gl / fp modules ----
  float* ugrow = allocF((size_t)BB * 64);
  gemm(gl, m_fp1.l[0], ugrow, BB, 1);
  float* up_gl = allocF((size_t)BB * NI * 64);
  k_broadcast<<<G1((size_t)BB * NI * 64), dim3(256), 0, stream>>>(ugrow, up_gl, NI, 64, BB);

  auto fpmod = [&](const float* kxyz, const float* kf, int Nk, int Ck, const Mlp& m, float* o) {
    k_interp3<<<G1((size_t)BB * NI), dim3(128), 0, stream>>>(x, kxyz, kf, T1, NI, Nk, Ck, BB);
    gemm(T1, m.l[0], o, (size_t)BB * NI, 1);
  };
  float* up_l5 = allocF((size_t)BB * NI * 64); fpmod(l5_xyz, l5_f, 128,  1024, m_fp2, up_l5);
  float* up_l4 = allocF((size_t)BB * NI * 64); fpmod(l4_xyz, l4_f, 256,  512,  m_fp3, up_l4);
  float* up_l3 = allocF((size_t)BB * NI * 64); fpmod(l3_xyz, l3_f, 512,  256,  m_fp4, up_l3);
  float* up_l2 = allocF((size_t)BB * NI * 64); fpmod(l2_xyz, l2_f, 1024, 128,  m_fp5, up_l2);

  // ---- 8 branches ----
  float* allf  = allocF((size_t)BB * NF * 128);
  float* xtbuf = allocF((size_t)BB * NI * 3);
  for (int i = 0; i < 8; ++i) {
    long long R = (long long)BB * NI;
    if (i > 3) {
      float* hc = run_mlp(m_stnc[i - 4], x, (size_t)BB * NI, T2, T3, 1);
      float* hmax = allocF((size_t)BB * 1024);
      k_maxpool<<<G1((size_t)BB * 1024), dim3(256), 0, stream>>>(hc, hmax, BB, NI, 1024);
      float* s1 = allocF((size_t)BB * 512);
      float* s2 = allocF((size_t)BB * 512);
      float* Tr = run_mlp(m_stnf[i - 4], hmax, (size_t)BB, s1, s2, 0);
      k_xt<<<G1((size_t)BB * NI * 3), dim3(256), 0, stream>>>(x, Tr, xtbuf, NI, BB);
    }
    k_copy_cols<<<G1((size_t)R * 64), dim3(256), 0, stream>>>(T1, 387, 0, up_gl, 64, R);
    if (i <= 3) {
      k_copy_cols<<<G1((size_t)R * 64), dim3(256), 0, stream>>>(T1, 387, 64,  up_l5, 64, R);
      k_copy_cols<<<G1((size_t)R * 64), dim3(256), 0, stream>>>(T1, 387, 128, up_l4, 64, R);
      k_copy_cols<<<G1((size_t)R * 64), dim3(256), 0, stream>>>(T1, 387, 192, up_l3, 64, R);
      k_copy_cols<<<G1((size_t)R * 64), dim3(256), 0, stream>>>(T1, 387, 256, up_l2, 64, R);
      k_copy_cols<<<G1((size_t)R * 64), dim3(256), 0, stream>>>(T1, 387, 320, l1_f,  64, R);
      k_copy_cols<<<G1((size_t)R * 3),  dim3(256), 0, stream>>>(T1, 387, 384, x,      3, R);
    } else {
      k_diff_cols<<<G1((size_t)R * 64), dim3(256), 0, stream>>>(T1, 387, 64,  up_gl, up_l5, 64, R);
      k_diff_cols<<<G1((size_t)R * 64), dim3(256), 0, stream>>>(T1, 387, 128, up_gl, up_l4, 64, R);
      k_diff_cols<<<G1((size_t)R * 64), dim3(256), 0, stream>>>(T1, 387, 192, up_gl, up_l3, 64, R);
      k_diff_cols<<<G1((size_t)R * 64), dim3(256), 0, stream>>>(T1, 387, 256, up_gl, up_l2, 64, R);
      k_diff_cols<<<G1((size_t)R * 64), dim3(256), 0, stream>>>(T1, 387, 320, up_gl, l1_f,  64, R);
      k_copy_cols<<<G1((size_t)R * 3),  dim3(256), 0, stream>>>(T1, 387, 384, xtbuf,  3, R);
    }
    float* res = run_mlp(m_mlps[i], T1, (size_t)R, T2, T3, 1);
    k_scatter_branch<<<G1((size_t)R * 128), dim3(256), 0, stream>>>(res, allf, i, NI, 128, BB);
  }

  // ---- coord / coord_feat ----
  float* coord_feat = allocF((size_t)BB * NF * 64);
  gemm(allf, m_sm1.l[0], coord_feat, (size_t)BB * NF, 1);
  float* coord = allocF((size_t)BB * NF * 3);
  gemm(coord_feat, m_sm2.l[0], coord, (size_t)BB * NF, 0);
  float* outp = (float*)d_out;
  k_copyf<<<G1((size_t)BB * NF * 3), dim3(256), 0, stream>>>(coord, outp, (long long)BB * NF * 3);

  // ---- FPS on coord, gather, score, topk ----
  const int NH = NF / 2; // 8192
  int* fi2 = allocI((size_t)BB * NH);
  float* dw2 = allocF((size_t)BB * NF);
  k_fps<<<dim3(BB), dim3(256), 0, stream>>>(coord, fi2, dw2, NF, NH);
  float* feat_fps  = allocF((size_t)BB * NH * 64);
  float* coord_fps = allocF((size_t)BB * NH * 3);
  k_gather_rows<<<G1((size_t)BB * NH * 64), dim3(256), 0, stream>>>(coord_feat, fi2, feat_fps, NF, NH, 64, BB);
  k_gather_rows<<<G1((size_t)BB * NH * 3),  dim3(256), 0, stream>>>(coord,      fi2, coord_fps, NF, NH, 3,  BB);

  float* sc = run_mlp(m_score, feat_fps, (size_t)BB * NH, T1, T2, 1);
  float* score = allocF((size_t)BB * NH);
  k_softplus<<<G1((size_t)BB * NH), dim3(256), 0, stream>>>(sc, score, (long long)BB * NH);

  int* tix = allocI((size_t)BB * NC);
  k_topk<<<dim3(BB), dim3(1024), (size_t)NH * 8, stream>>>(score, tix, NH, NC);

  float* coarse = allocF((size_t)BB * NC * 3);
  float* cfeat  = allocF((size_t)BB * NC * 64);
  k_gather_rows<<<G1((size_t)BB * NC * 3),  dim3(256), 0, stream>>>(coord_fps, tix, coarse, NH, NC, 3,  BB);
  k_gather_rows<<<G1((size_t)BB * NC * 64), dim3(256), 0, stream>>>(feat_fps,  tix, cfeat,  NH, NC, 64, BB);
  k_copyf<<<G1((size_t)BB * NC * 3), dim3(256), 0, stream>>>(coarse, outp + (size_t)BB * NF * 3,
                                                             (long long)BB * NC * 3);

  // ---- fine ----
  float* center  = allocF((size_t)BB * NF * 3);
  float* cfeat_r = allocF((size_t)BB * NF * 64);
  k_repeat<<<G1((size_t)BB * NF * 3),  dim3(256), 0, stream>>>(coarse, center,  NC, 4, 3,  BB);
  k_repeat<<<G1((size_t)BB * NF * 64), dim3(256), 0, stream>>>(cfeat,  cfeat_r, NC, 4, 64, BB);

  float* fin = allocF((size_t)BB * NF * 1093);
  k_build_fin<<<G1((size_t)BB * NF * 1093), dim3(256), 0, stream>>>(fin, center, cfeat_r, gl, NF, BB);

  gemm(fin, m_final.l[0], T1, (size_t)BB * NF, 1);
  gemm(T1,  m_final.l[1], T2, (size_t)BB * NF, 1);
  float* delta = allocF((size_t)BB * NF * 3);
  gemm(T2,  m_final.l[2], delta, (size_t)BB * NF, 0);
  k_addf<<<G1((size_t)BB * NF * 3), dim3(256), 0, stream>>>(
      center, delta, outp + (size_t)BB * NF * 3 + (size_t)BB * NC * 3, (long long)BB * NF * 3);
}
